// ARCQuantLayer_1803886264481
// MI455X (gfx1250) — 16-bit and 8-bit matrix operand order repaired and checked by reading, compiled, not run on hardware
//
#include <hip/hip_runtime.h>

typedef int      v8i  __attribute__((ext_vector_type(8)));
typedef float    v8f  __attribute__((ext_vector_type(8)));
typedef _Float16 v16h __attribute__((ext_vector_type(16)));

#define K_DIM 4096
#define N_DIM 4096
#define KO    256
#define BM    64
#define BN    256
#define BK    64
#define LDSS  80
#define A_BYTES   (BM * LDSS)
#define B_BYTES   (BN * LDSS)
#define BUF_BYTES (A_BYTES + B_BYTES)


__global__ void quant8_kernel(const float* __restrict__ x, signed char* __restrict__ q, int n4) {
    int i = blockIdx.x * blockDim.x + threadIdx.x;
    if (i >= n4) return;
    float4 v = ((const float4*)x)[i];
    int a = __float2int_rn(v.x * 8.0f);
    int b = __float2int_rn(v.y * 8.0f);
    int c = __float2int_rn(v.z * 8.0f);
    int d = __float2int_rn(v.w * 8.0f);
    a = min(max(a, -127), 127); b = min(max(b, -127), 127);
    c = min(max(c, -127), 127); d = min(max(d, -127), 127);
    char4 o; o.x = (signed char)a; o.y = (signed char)b; o.z = (signed char)c; o.w = (signed char)d;
    ((char4*)q)[i] = o;
}

__global__ void resid_kernel(const float* __restrict__ x, const int* __restrict__ idx,
                             _Float16* __restrict__ xres, int M, int nOut) {
    int i = blockIdx.x * blockDim.x + threadIdx.x;
    int row = i >> 8;
    int j   = i & (KO - 1);
    if (row >= M) return;
    float r = 0.0f;
    if (j < nOut) {
        int d = idx[j];
        float xv = x[(size_t)row * K_DIM + d];
        float q  = (float)__float2int_rn(xv * 8.0f) * 0.125f;
        r = xv - q;
    }
    xres[(size_t)row * KO + j] = (_Float16)r;
}

__global__ void arcpad_kernel(const float* __restrict__ arc, _Float16* __restrict__ arcw,
                              int N, int nOut) {
    int i = blockIdx.x * blockDim.x + threadIdx.x;
    int o = i >> 8;
    int j = i & (KO - 1);
    if (o >= N) return;
    float v = (j < nOut) ? arc[(size_t)o * nOut + j] : 0.0f;
    arcw[(size_t)o * KO + j] = (_Float16)v;
}


__device__ __forceinline__ void async_ld_b128(unsigned lds_off, const void* gptr) {
    asm volatile("global_load_async_to_lds_b128 %0, %1, off"
                 :: "v"(lds_off), "v"(gptr)
                 : "memory");
}

__device__ __forceinline__ void wait_async0() {
    asm volatile("s_wait_asynccnt 0" ::: "memory");
}


__global__ __launch_bounds__(256, 2)
void arc_gemm_kernel(const signed char* __restrict__ xq,
                     const signed char* __restrict__ wq,
                     const _Float16* __restrict__ xres,
                     const _Float16* __restrict__ arcw,
                     float* __restrict__ out) {
    __shared__ __attribute__((aligned(16))) signed char smem[2][BUF_BYTES];

    const int tid  = threadIdx.x;
    const int lane = tid & 31;
    const int wave = tid >> 5;
    const int half = lane >> 4;
    const int l16  = lane & 15;
    const int m0   = (wave >> 2) * 32;
    const int n0   = (wave & 3) * 64;
    const int blockM = blockIdx.y * BM;
    const int blockN = blockIdx.x * BN;

    const int ldRow = tid >> 2;
    const int ldCol = (tid & 3) * 16;

    const unsigned lbase0 = (unsigned)(size_t)(&smem[0][0]);
    const unsigned lbase1 = (unsigned)(size_t)(&smem[1][0]);
    const unsigned aOff = (unsigned)(ldRow * LDSS + ldCol);
    const unsigned bOff = (unsigned)(A_BYTES + ldRow * LDSS + ldCol);

    const signed char* gA = xq + (size_t)(blockM + ldRow) * K_DIM + ldCol;
    const signed char* gB = wq + (size_t)(blockN + ldRow) * K_DIM + ldCol;

    auto stage = [&](unsigned lbase, int k0) {
        async_ld_b128(lbase + aOff, gA + k0);
        async_ld_b128(lbase + bOff,               gB + k0);
        async_ld_b128(lbase + bOff + 64u * LDSS,  gB + (size_t)64  * K_DIM + k0);
        async_ld_b128(lbase + bOff + 128u * LDSS, gB + (size_t)128 * K_DIM + k0);
        async_ld_b128(lbase + bOff + 192u * LDSS, gB + (size_t)192 * K_DIM + k0);
    };

    v8i acc[2][4] = {};

    auto compute = [&](int buf) {
        const signed char* sA = &smem[buf][0];
        const signed char* sB = &smem[buf][A_BYTES];
        v8i aF[2], bF[4];
#pragma unroll
        for (int t = 0; t < 2; ++t) {
            const signed char* pa = &sA[(m0 + t * 16 + l16) * LDSS + 8 * half];
            int2 a0 = *(const int2*)(pa + 0);
            int2 a1 = *(const int2*)(pa + 16);
            int2 a2 = *(const int2*)(pa + 32);
            int2 a3 = *(const int2*)(pa + 48);
            v8i a;
            a[0] = a0.x; a[1] = a0.y; a[2] = a1.x; a[3] = a1.y;
            a[4] = a2.x; a[5] = a2.y; a[6] = a3.x; a[7] = a3.y;
            aF[t] = a;
        }
#pragma unroll
        for (int t = 0; t < 4; ++t) {
            const signed char* pb = &sB[(n0 + t * 16 + l16) * LDSS + 8 * half];
            int2 b0 = *(const int2*)(pb + 0);
            int2 b1 = *(const int2*)(pb + 16);
            int2 b2 = *(const int2*)(pb + 32);
            int2 b3 = *(const int2*)(pb + 48);
            v8i b;
            b[0] = b0.x; b[1] = b0.y; b[2] = b1.x; b[3] = b1.y;
            b[4] = b2.x; b[5] = b2.y; b[6] = b3.x; b[7] = b3.y;
            bF[t] = b;
        }
#pragma unroll
        for (int i = 0; i < 2; ++i)
#pragma unroll
            for (int j = 0; j < 4; ++j)
                acc[i][j] = __builtin_amdgcn_wmma_i32_16x16x64_iu8(
                    true, aF[i], true, bF[j], acc[i][j], false, false);
    };

    stage(lbase0, 0);
    for (int k0 = 0; k0 < K_DIM; k0 += 2 * BK) {
        wait_async0();
        __syncthreads();
        if (k0 + BK < K_DIM) stage(lbase1, k0 + BK);
        compute(0);
        wait_async0();
        __syncthreads();
        if (k0 + 2 * BK < K_DIM) stage(lbase0, k0 + 2 * BK);
        compute(1);
    }

    v8f facc[2][4];
#pragma unroll
    for (int i = 0; i < 2; ++i)
#pragma unroll
        for (int j = 0; j < 4; ++j)
#pragma unroll
            for (int e = 0; e < 8; ++e)
                facc[i][j][e] = (float)acc[i][j][e] * 0.015625f;

    for (int kk = 0; kk < KO; kk += 32) {
        v16h aH[2], bH[4];
#pragma unroll
        for (int t = 0; t < 2; ++t) {
            const _Float16* p = xres + (size_t)(blockM + m0 + t * 16 + l16) * KO + kk + 8 * half;
            union { uint4 u[2]; v16h h; } ca;
            ca.u[0] = *(const uint4*)(p);
            ca.u[1] = *(const uint4*)(p + 16);
            aH[t] = ca.h;
        }
#pragma unroll
        for (int t = 0; t < 4; ++t) {
            const _Float16* q = arcw + (size_t)(blockN + n0 + t * 16 + l16) * KO + kk + 8 * half;
            union { uint4 u[2]; v16h h; } cb;
            cb.u[0] = *(const uint4*)(q);
            cb.u[1] = *(const uint4*)(q + 16);
            bH[t] = cb.h;
        }
#pragma unroll
        for (int i = 0; i < 2; ++i)
#pragma unroll
            for (int j = 0; j < 4; ++j)
                facc[i][j] = __builtin_amdgcn_wmma_f32_16x16x32_f16(
                    false, aH[i], false, bH[j], (short)0, facc[i][j], false, false);
    }

#pragma unroll
    for (int i = 0; i < 2; ++i)
#pragma unroll
        for (int j = 0; j < 4; ++j) {
            const size_t rowBase = (size_t)(blockM + m0 + i * 16 + 8 * half);
            const int col = blockN + n0 + j * 16 + l16;
#pragma unroll
            for (int r = 0; r < 8; ++r)
                out[(rowBase + r) * (size_t)N_DIM + col] = facc[i][j][r];
        }
}


extern "C" void kernel_launch(void* const* d_in, const int* in_sizes, int n_in,
                              void* d_out, int out_size, void* d_ws, size_t ws_size,
                              hipStream_t stream) {
    const float* x   = (const float*)d_in[0];
    const float* w   = (const float*)d_in[1];
    const float* arc = (const float*)d_in[2];
    const int*   idx = (const int*)d_in[3];
    float* out = (float*)d_out;

    const int nOut = in_sizes[3];
    const int M    = in_sizes[0] / K_DIM;

    signed char* xq = (signed char*)d_ws;
    signed char* wq = xq + (size_t)M * K_DIM;
    _Float16* xres  = (_Float16*)(wq + (size_t)N_DIM * K_DIM);
    _Float16* arcw  = xres + (size_t)M * KO;

    int n4x = M * K_DIM / 4;
    quant8_kernel<<<(n4x + 255) / 256, 256, 0, stream>>>(x, xq, n4x);
    int n4w = N_DIM * K_DIM / 4;
    quant8_kernel<<<(n4w + 255) / 256, 256, 0, stream>>>(w, wq, n4w);
    int nr = M * KO;
    resid_kernel<<<(nr + 255) / 256, 256, 0, stream>>>(x, idx, xres, M, nOut);
    int na = N_DIM * KO;
    arcpad_kernel<<<(na + 255) / 256, 256, 0, stream>>>(arc, arcw, N_DIM, nOut);

    dim3 grid(N_DIM / BN, M / BM);
    arc_gemm_kernel<<<grid, 256, 0, stream>>>(xq, wq, xres, arcw, out);
}
